// GCNLSTMEncoder_83288005804581
// MI455X (gfx1250) — compile-verified, matrix operand order checked by reading, not run on hardware
//
#include <hip/hip_runtime.h>
#include <hip/hip_bf16.h>
#include <math.h>
#include <stdint.h>

// ---------------------------------------------------------------------------
// GCN + LSTM encoder for MI455X (gfx1250, wave32, WMMA + TDM).
// B=16, N=64, T=128, D=8, N_EMB=64, FEAT=128, N_H=128, GATES=512, BN=1024.
// ---------------------------------------------------------------------------

typedef __attribute__((ext_vector_type(16))) __bf16        v16bf;
typedef __attribute__((ext_vector_type(8)))  __bf16        v8bf;
typedef __attribute__((ext_vector_type(8)))  float         v8f;
typedef __attribute__((ext_vector_type(4)))  unsigned int  u32x4;
typedef __attribute__((ext_vector_type(8)))  int           i32x8;
typedef __attribute__((ext_vector_type(4)))  int           i32x4;

union BFrag { v16bf v; v8bf h[2]; };

#define T_STEPS 128
#define NNODES  64
#define DDIM    8
#define NEMB    64
#define FEAT    128
#define NH      128
#define NGATES  512
#define BN      1024

// workspace layout (bytes)
#define WS_XSEQ_OFF   0u
#define WS_XSEQ_BYTES ((size_t)T_STEPS * BN * FEAT * 2)        // 32 MB bf16
#define WS_WIH_OFF    (WS_XSEQ_BYTES)                          // 128 KB bf16
#define WS_WHH_OFF    (WS_WIH_OFF + (size_t)NGATES * FEAT * 2) // 128 KB bf16
#define WS_BSUM_OFF   (WS_WHH_OFF + (size_t)NGATES * NH * 2)   //   2 KB f32

#if __has_builtin(__builtin_amdgcn_tensor_load_to_lds) && \
    __has_builtin(__builtin_amdgcn_s_wait_tensorcnt)
#define USE_TDM 1
#endif

#ifdef USE_TDM
// Issue a TDM DMA of a contiguous [rows x cols] bf16 tile from global into LDS.
// D# per CDNA5 ISA ch.8: group0 = count/lds_addr/global_addr/type,
// group1 = data_size + tensor/tile dims + dim0 stride, groups 2/3 = 0 (2D).
__device__ __forceinline__ void tdm_load_2d_bf16(unsigned lds_off,
                                                 const void* gptr,
                                                 unsigned rows, unsigned cols) {
    unsigned long long ga = (unsigned long long)(uintptr_t)gptr;
    u32x4 g0;
    g0[0] = 1u;                                          // count=1 (valid D#)
    g0[1] = lds_off;                                     // lds_addr (bytes)
    g0[2] = (unsigned)(ga & 0xFFFFFFFFu);                // global_addr[31:0]
    g0[3] = (unsigned)((ga >> 32) & 0x01FFFFFFu) | (2u << 30); // [56:32] | type=2
    i32x8 g1;
    g1[0] = (int)(1u << 16);                             // data_size=1 (2 bytes)
    g1[1] = (int)((cols & 0xFFFFu) << 16);               // tensor_dim0[15:0]
    g1[2] = (int)((cols >> 16) | ((rows & 0xFFFFu) << 16));   // dim0 hi | dim1 lo
    g1[3] = (int)((rows >> 16) | ((cols & 0xFFFFu) << 16));   // dim1 hi | tile_dim0
    g1[4] = (int)(rows & 0xFFFFu);                       // tile_dim1 (tile_dim2=0)
    g1[5] = (int)cols;                                   // tensor_dim0_stride lo32
    g1[6] = 0;                                           // stride0 hi | stride1 lo
    g1[7] = 0;                                           // stride1 hi
    i32x4 gz = {0, 0, 0, 0};
#if __has_include(<hip/amd_detail/amd_gfx1250_TDM.h>)
    i32x8 gz8 = {0, 0, 0, 0, 0, 0, 0, 0};
    __builtin_amdgcn_tensor_load_to_lds(g0, g1, gz, gz, gz8, 0);
#else
    __builtin_amdgcn_tensor_load_to_lds(g0, g1, gz, gz, 0);
#endif
}
#endif

// ---------------------------------------------------------------------------
// Kernel 0: weight prep. fp32 W_ih/W_hh -> bf16 in workspace (TDM source),
// combined bias b_ih + b_hh -> f32 in workspace.
// ---------------------------------------------------------------------------
__global__ __launch_bounds__(256)
void prep_kernel(const float* __restrict__ Wih, const float* __restrict__ Whh,
                 const float* __restrict__ bih, const float* __restrict__ bhh,
                 unsigned short* __restrict__ wih_b,
                 unsigned short* __restrict__ whh_b,
                 float* __restrict__ bsum)
{
    int i = blockIdx.x * 256 + threadIdx.x;          // 65536 threads total
    if (i < NGATES * FEAT) {
        __bf16 a = (__bf16)Wih[i];
        __bf16 b = (__bf16)Whh[i];
        wih_b[i] = __builtin_bit_cast(unsigned short, a);
        whh_b[i] = __builtin_bit_cast(unsigned short, b);
    }
    if (i < NGATES) bsum[i] = bih[i] + bhh[i];
}

// ---------------------------------------------------------------------------
// Kernel 1: GCN front-end. One workgroup per (b, t). Produces the LSTM input
// sequence x_seq[t][bn][f] in bf16 (f: 0..63 = selu(skip), 64..127 = selu(conv)).
// ---------------------------------------------------------------------------
__global__ __launch_bounds__(256)
void gcn_kernel(const float* __restrict__ x,
                const float* __restrict__ Wc, const float* __restrict__ bc,
                const float* __restrict__ Ws, const float* __restrict__ bs,
                unsigned short* __restrict__ xseq_u)
{
    __bf16* xseq = reinterpret_cast<__bf16*>(xseq_u);

    const int bt  = blockIdx.x;        // 0 .. B*T-1
    const int b   = bt >> 7;           // T = 128
    const int t   = bt & 127;
    const int tid = threadIdx.x;

    __shared__ float sx[NNODES][DDIM];      // x[b, :, t, :]
    __shared__ float sA[NNODES][NNODES];    // inv-distance adjacency
    __shared__ float srow[NNODES];          // 1 / rowsum
    __shared__ float sxc[NNODES][DDIM];     // A_norm @ x

    for (int i = tid; i < NNODES * DDIM; i += 256) {
        int n = i >> 3, d = i & 7;
        sx[n][d] = x[(((size_t)(b * NNODES + n)) * T_STEPS + t) * DDIM + d];
    }
    __syncthreads();

    // pairwise inverse distances (diagonal = 0, fully-connected off-diagonal)
    for (int i = tid; i < NNODES * NNODES; i += 256) {
        int s = i >> 6, r = i & 63;
        float acc = 0.f;
#pragma unroll
        for (int d = 0; d < DDIM; ++d) {
            float df = sx[r][d] - sx[s][d];
            acc += df * df;
        }
        sA[s][r] = (r == s) ? 0.f : 1.f / (sqrtf(acc) + 1e-20f);
    }
    __syncthreads();

    if (tid < NNODES) {
        float rs = 0.f;
        for (int r = 0; r < NNODES; ++r) rs += sA[tid][r];
        srow[tid] = 1.f / fmaxf(rs, 1e-12f);
    }
    __syncthreads();

    // x_conv = row_normalize(A) @ x
    for (int i = tid; i < NNODES * DDIM; i += 256) {
        int n = i >> 3, d = i & 7;
        float acc = 0.f;
        for (int m = 0; m < NNODES; ++m) acc += sA[n][m] * sx[m][d];
        sxc[n][d] = acc * srow[n];
    }
    __syncthreads();

    // projections + selu + bf16 store into x_seq[t][b*64+n][f]
    const float kSeluScale = 1.0507009873554805f;
    const float kSeluAlpha = 1.6732632423543772f;
    for (int i = tid; i < NNODES * FEAT; i += 256) {
        int n = i >> 7, f = i & 127;
        float acc;
        if (f < NEMB) {                       // skip branch: x @ Ws^T + bs
            int e = f;
            acc = bs[e];
#pragma unroll
            for (int d = 0; d < DDIM; ++d) acc += sx[n][d] * Ws[e * DDIM + d];
        } else {                              // conv branch: x_conv @ Wc^T + bc
            int e = f - NEMB;
            acc = bc[e];
#pragma unroll
            for (int d = 0; d < DDIM; ++d) acc += sxc[n][d] * Wc[e * DDIM + d];
        }
        float r = (acc > 0.f) ? acc : kSeluAlpha * (__expf(acc) - 1.f);
        r *= kSeluScale;
        xseq[((size_t)t * BN + (b * NNODES + n)) * FEAT + f] = (__bf16)r;
    }
}

// ---------------------------------------------------------------------------
// Kernel 2: persistent LSTM. 64 workgroups x 256 threads (8 waves).
// Each workgroup owns 16 batch rows for all 128 timesteps. Both weight
// matrices are DMA'd into LDS once via the Tensor Data Mover (bf16,
// 2 x 128 KB of CDNA5's 320 KB WGP LDS) and reused for every step.
// Each wave computes a 16-row x 64-gate-column stripe with
// v_wmma_f32_16x16x32_bf16; accumulators start from inline-0 C and the
// combined bias is folded into the gate-staging epilogue.
// ---------------------------------------------------------------------------
__global__ __launch_bounds__(256)
void lstm_kernel(const unsigned short* __restrict__ xseq_u,
                 const unsigned short* __restrict__ wih_u,
                 const unsigned short* __restrict__ whh_u,
                 const float* __restrict__ bsum,
                 float* __restrict__ out)
{
    const __bf16* xseq = reinterpret_cast<const __bf16*>(xseq_u);

    __shared__ __bf16 sWih[NGATES][FEAT];    // 128 KB
    __shared__ __bf16 sWhh[NGATES][NH];      // 128 KB
    __shared__ float  sGates[16][NGATES];    //  32 KB
    __shared__ float  sC[16][NH];            //   8 KB
    __shared__ __bf16 sH[16][NH];            //   4 KB
    __shared__ float  sBias[NGATES];         //   2 KB

    const int tid     = threadIdx.x;
    const int wave    = tid >> 5;
    const int lane    = tid & 31;
    const int laneRow = lane & 15;   // M row (A) / N col (B) within a tile
    const int laneHi  = lane >> 4;   // half-wave selector for K grouping
    const int rowBase = blockIdx.x * 16;
    const int n0base  = wave * 64;   // this wave's first gate column

#ifdef USE_TDM
    // one TDM DMA per weight matrix, issued by one wave each (TDM ignores EXEC)
    if (wave == 0) {
        tdm_load_2d_bf16((unsigned)(uintptr_t)&sWih[0][0], wih_u, NGATES, FEAT);
        __builtin_amdgcn_s_wait_tensorcnt(0);
    } else if (wave == 1) {
        tdm_load_2d_bf16((unsigned)(uintptr_t)&sWhh[0][0], whh_u, NGATES, NH);
        __builtin_amdgcn_s_wait_tensorcnt(0);
    }
#else
    {
        const __bf16* wih = reinterpret_cast<const __bf16*>(wih_u);
        const __bf16* whh = reinterpret_cast<const __bf16*>(whh_u);
        for (int i = tid; i < NGATES * FEAT; i += 256) (&sWih[0][0])[i] = wih[i];
        for (int i = tid; i < NGATES * NH;  i += 256) (&sWhh[0][0])[i] = whh[i];
    }
#endif
    for (int i = tid; i < NGATES; i += 256) sBias[i] = bsum[i];
    for (int i = tid; i < 16 * NH; i += 256) {
        (&sC[0][0])[i] = 0.f;
        (&sH[0][0])[i] = (__bf16)0.f;
    }
    __syncthreads();

    for (int t = 0; t < T_STEPS; ++t) {
        // ---- A fragments: x_t rows from global (bf16, 16B chunks) ----
        const __bf16* xrow =
            xseq + ((size_t)t * BN + rowBase + laneRow) * FEAT;
        // speculative prefetch of next timestep's activations (global_prefetch)
        __builtin_prefetch(xrow + (size_t)BN * FEAT, 0, 1);

        BFrag ax[4], ah[4];
        const __bf16* hrow = &sH[laneRow][0];
#pragma unroll
        for (int kc = 0; kc < 4; ++kc) {
            int kb = kc * 32 + laneHi * 8;
            ax[kc].h[0] = *(const v8bf*)(xrow + kb);
            ax[kc].h[1] = *(const v8bf*)(xrow + kb + 16);
            ah[kc].h[0] = *(const v8bf*)(hrow + kb);
            ah[kc].h[1] = *(const v8bf*)(hrow + kb + 16);
        }

        // ---- accumulators: zero (first WMMA uses inline-0 SRC2) ----
        v8f acc[4];
#pragma unroll
        for (int i = 0; i < 4; ++i)
#pragma unroll
            for (int r = 0; r < 8; ++r) acc[i][r] = 0.f;

        // ---- WMMA: 4 N-tiles x 4 K-chunks x (input + hidden) ----
#pragma unroll
        for (int i = 0; i < 4; ++i) {
            int n = n0base + i * 16 + laneRow;      // B column for this lane
            const __bf16* wihrow = &sWih[n][0];
            const __bf16* whhrow = &sWhh[n][0];
#pragma unroll
            for (int kc = 0; kc < 4; ++kc) {
                int kb = kc * 32 + laneHi * 8;
                BFrag bw, bh;
                bw.h[0] = *(const v8bf*)(wihrow + kb);
                bw.h[1] = *(const v8bf*)(wihrow + kb + 16);
                acc[i] = __builtin_amdgcn_wmma_f32_16x16x32_bf16(
                    false, ax[kc].v, false, bw.v, (short)0, acc[i],
                    false, false);
                bh.h[0] = *(const v8bf*)(whhrow + kb);
                bh.h[1] = *(const v8bf*)(whhrow + kb + 16);
                acc[i] = __builtin_amdgcn_wmma_f32_16x16x32_bf16(
                    false, ah[kc].v, false, bh.v, (short)0, acc[i],
                    false, false);
            }
        }

        // ---- stage gates + bias to LDS (VGPR r -> row r + 8*laneHi) ----
#pragma unroll
        for (int i = 0; i < 4; ++i) {
            int col = n0base + i * 16 + laneRow;
            float bv = sBias[col];
#pragma unroll
            for (int r = 0; r < 8; ++r)
                sGates[r + laneHi * 8][col] = acc[i][r] + bv;
        }
        __syncthreads();

        // ---- fp32 cell update: 16x128 cells, 8 per thread ----
#pragma unroll
        for (int u = 0; u < 8; ++u) {
            int idx = tid + u * 256;          // 0 .. 2047
            int rr  = idx >> 7, cc = idx & 127;
            float ig = sGates[rr][cc];
            float fg = sGates[rr][NH + cc];
            float gg = sGates[rr][2 * NH + cc];
            float og = sGates[rr][3 * NH + cc];
            float si = 1.f / (1.f + __expf(-ig));
            float sf = 1.f / (1.f + __expf(-fg));
            float so = 1.f / (1.f + __expf(-og));
            float c  = sf * sC[rr][cc] + si * tanhf(gg);
            float h  = so * tanhf(c);
            sC[rr][cc] = c;
            sH[rr][cc] = (__bf16)h;
            if (t == T_STEPS - 1)
                out[(size_t)(rowBase + rr) * NH + cc] = h;
        }
        __syncthreads();
    }
}

// ---------------------------------------------------------------------------
// Launch: prep (bf16 weights + bias), GCN (2048 WGs), persistent LSTM (64 WGs).
// Workspace: x_seq bf16 32 MB, then bf16 weights (2 x 128 KB), then bias.
// ---------------------------------------------------------------------------
extern "C" void kernel_launch(void* const* d_in, const int* in_sizes, int n_in,
                              void* d_out, int out_size, void* d_ws, size_t ws_size,
                              hipStream_t stream) {
    (void)in_sizes; (void)n_in; (void)out_size; (void)ws_size;
    const float* x    = (const float*)d_in[0];
    // d_in[1] = rel_rec, d_in[2] = rel_send: fully-connected structure is
    // known statically, so they are not needed.
    const float* Wc   = (const float*)d_in[3];
    const float* bc   = (const float*)d_in[4];
    const float* Ws   = (const float*)d_in[5];
    const float* bs   = (const float*)d_in[6];
    const float* Wih  = (const float*)d_in[7];
    const float* Whh  = (const float*)d_in[8];
    const float* bih  = (const float*)d_in[9];
    const float* bhh  = (const float*)d_in[10];

    char* ws = (char*)d_ws;
    unsigned short* xseq  = (unsigned short*)(ws + WS_XSEQ_OFF);
    unsigned short* wih_b = (unsigned short*)(ws + WS_WIH_OFF);
    unsigned short* whh_b = (unsigned short*)(ws + WS_WHH_OFF);
    float*          bsum  = (float*)(ws + WS_BSUM_OFF);
    float* out = (float*)d_out;

    prep_kernel<<<256, 256, 0, stream>>>(Wih, Whh, bih, bhh, wih_b, whh_b, bsum);
    gcn_kernel<<<16 * T_STEPS, 256, 0, stream>>>(x, Wc, bc, Ws, bs, xseq);
    lstm_kernel<<<BN / 16, 256, 0, stream>>>(xseq, wih_b, whh_b, bsum, out);
}
